// BasicModelNormExtraLarge_12300786336359
// MI455X (gfx1250) — matrix operand order repaired and checked by reading, not compiled, not run on hardware
//
#include <hip/hip_runtime.h>

#define N_NODES 50000
#define N_EDGES 1600000

typedef __attribute__((ext_vector_type(16))) __bf16 v16bf;
typedef __attribute__((ext_vector_type(8)))  float  v8f;

__device__ __forceinline__ __bf16 f2bf(float f) {
  // round-to-nearest-even fp32 -> bf16
  unsigned u = __builtin_bit_cast(unsigned, f);
  u += 0x7FFFu + ((u >> 16) & 1u);
  unsigned short h = (unsigned short)(u >> 16);
  return __builtin_bit_cast(__bf16, h);
}

// ---------------------------------------------------------------------------
// Pack W[K,F] fp32 -> bf16 tiles in exact WMMA B-fragment order.
// Tile (kt, nt) = 32 lanes x 16 bf16 (1 KB). Lane holds column n = nt*16 +
// (lane&15), k = kt*32 + (lane>=16 ? 8 : 0) + (e&7) + (e>=8 ? 16 : 0). k >= K -> 0.
// ---------------------------------------------------------------------------
__global__ void pack_w(const float* __restrict__ W, __bf16* __restrict__ Wp,
                       int K, int F, long long total) {
  long long idx = (long long)blockIdx.x * blockDim.x + threadIdx.x;
  if (idx >= total) return;
  const int e    = (int)(idx & 15);
  const int lane = (int)((idx >> 4) & 31);
  const long long tile = idx >> 9;
  const int nT = F >> 4;
  const int nt = (int)(tile % nT);
  const int kt = (int)(tile / nT);
  const int n = nt * 16 + (lane & 15);
  const int k = kt * 32 + ((lane >= 16) ? 8 : 0) + (e & 7) + ((e & 8) ? 16 : 0);
  const float v = (k < K) ? W[(size_t)k * F + n] : 0.f;
  Wp[idx] = f2bf(v);
}

// ---------------------------------------------------------------------------
// Convert activations fp32 [rows,K] -> bf16 [rows, Kp] row-major (Kp = 1<<lgKp,
// zero-padded), with optional fused ReLU.
// ---------------------------------------------------------------------------
template <bool RELU>
__global__ void cvt_a(const float* __restrict__ A, __bf16* __restrict__ Ab,
                      long long total, int K, int lgKp) {
  long long idx = (long long)blockIdx.x * blockDim.x + threadIdx.x;
  if (idx >= total) return;
  const int k = (int)(idx & ((1 << lgKp) - 1));
  const long long r = idx >> lgKp;
  float v = (k < K) ? A[r * (long long)K + k] : 0.f;
  if (RELU) v = fmaxf(v, 0.f);
  Ab[idx] = f2bf(v);
}

// ---------------------------------------------------------------------------
// WMMA GEMM: C[N,F] = Ab[N,Kp](bf16) @ Wp(packed bf16 tiles)
// block = 256 threads = 8 waves; wave -> 32 x (NT*16) strip (2 A frags x NT B
// frags, 2*NT WMMAs per 32-wide K step); block -> 256 rows.
// NT is compile-time (4 for F in {4096,1024,256,64}; 1 for F=16) so the hot
// loop is completely branch-free. A rows clamped (garbage rows never stored)
// so EXEC stays all-ones for WMMA.
// ---------------------------------------------------------------------------
template <int NT>
__global__ __launch_bounds__(256) void gemm_wmma_bf16(
    const __bf16* __restrict__ Ab, const __bf16* __restrict__ Wp,
    float* __restrict__ C, int N, int Kt, int lgKp, int F) {
  const int lane  = threadIdx.x & 31;
  const int wave  = threadIdx.x >> 5;
  const int mBase = blockIdx.y * 256 + wave * 32;
  const int nBase = blockIdx.x * (NT * 16);
  if (mBase >= N) return;                               // wave-uniform
  const int nT16 = F >> 4;

  // A fragment pointers (rows clamped -> no per-lane predication)
  int m0 = mBase + (lane & 15);
  int m1 = m0 + 16;
  m0 = m0 < N - 1 ? m0 : N - 1;
  m1 = m1 < N - 1 ? m1 : N - 1;
  const int kA = (lane >> 4) ? 8 : 0;
  const __bf16* a0 = Ab + (((size_t)m0) << lgKp) + kA;
  const __bf16* a1 = Ab + (((size_t)m1) << lgKp) + kA;
  // B packed pointer: tile (kt=0, nt = nBase/16), this lane's 16 values
  const __bf16* wp = Wp + ((size_t)(nBase >> 4)) * 512 + lane * 16;
  const size_t wStep = (size_t)nT16 * 512;              // next kt

  v8f acc[2][NT];
#pragma unroll
  for (int t = 0; t < NT; ++t) { acc[0][t] = v8f{}; acc[1][t] = v8f{}; }

  for (int kt = 0; kt < Kt; ++kt) {
    union Frag { uint4 u[2]; v16bf v; };
    Frag af0, af1, bf[NT];
    // A: halfs [kA..kA+7] at byte 0, [kA+16..kA+23] at byte 32
    const uint4* pa0 = (const uint4*)a0;
    const uint4* pa1 = (const uint4*)a1;
    af0.u[0] = pa0[0]; af0.u[1] = pa0[2];
    af1.u[0] = pa1[0]; af1.u[1] = pa1[2];
    // B: NT tiles, 32 contiguous bytes per lane, 1 KB tile stride
    const uint4* pb = (const uint4*)wp;
#pragma unroll
    for (int t = 0; t < NT; ++t) {
      bf[t].u[0] = pb[t * 64];
      bf[t].u[1] = pb[t * 64 + 1];
    }
#pragma unroll
    for (int t = 0; t < NT; ++t) {
      acc[0][t] = __builtin_amdgcn_wmma_f32_16x16x32_bf16(
          false, af0.v, false, bf[t].v, (short)0, acc[0][t], false, false);
      acc[1][t] = __builtin_amdgcn_wmma_f32_16x16x32_bf16(
          false, af1.v, false, bf[t].v, (short)0, acc[1][t], false, false);
    }
    a0 += 32; a1 += 32; wp += wStep;
  }

  // store: VGPR r -> row mBase + g*16 + r + (lane>=16 ? 8 : 0), col = lane&15
  const int mOut = mBase + ((lane >> 4) ? 8 : 0);
#pragma unroll
  for (int g = 0; g < 2; ++g) {
#pragma unroll
    for (int t = 0; t < NT; ++t) {
      const int n = nBase + t * 16 + (lane & 15);
#pragma unroll
      for (int r = 0; r < 8; ++r) {
        const int m = mOut + g * 16 + r;
        if (m < N) C[(size_t)m * F + n] = acc[g][t][r];
      }
    }
  }
}

// ---------------------------------------------------------------------------
// GCN plumbing kernels
// ---------------------------------------------------------------------------
__global__ void deg_init(float* __restrict__ deg, int n) {
  int i = blockIdx.x * blockDim.x + threadIdx.x;
  if (i < n) deg[i] = 1.0f;  // self-loop
}
__global__ void deg_accum(const int* __restrict__ dst, float* __restrict__ deg, int E) {
  int e = blockIdx.x * blockDim.x + threadIdx.x;
  if (e < E) atomicAdd(&deg[dst[e]], 1.0f);
}
__global__ void deg_rsqrt(float* __restrict__ deg, int n) {
  int i = blockIdx.x * blockDim.x + threadIdx.x;
  if (i < n) deg[i] = rsqrtf(deg[i]);
}

// out[i, f..f+3] = h[i, f..f+3]*dinv[i]^2 + b[f..f+3]   (float4, F = 1<<lgF)
__global__ void self_init4(const float4* __restrict__ h, const float* __restrict__ dinv,
                           const float* __restrict__ b, float4* __restrict__ out,
                           long long total4, int lgF) {
  long long idx = (long long)blockIdx.x * blockDim.x + threadIdx.x;
  if (idx >= total4) return;
  const long long base = idx << 2;
  const int i = (int)(base >> lgF);
  const int f = (int)(base & ((1 << lgF) - 1));
  const float di = dinv[i];
  const float s = di * di;
  const float4 hv = h[idx];
  float4 o;
  o.x = hv.x * s + b[f];
  o.y = hv.y * s + b[f + 1];
  o.z = hv.z * s + b[f + 2];
  o.w = hv.w * s + b[f + 3];
  out[idx] = o;
}

// out[dst[e], f..f+3] += h[src[e], f..f+3] * dinv[src]*dinv[dst]
// consecutive threads = consecutive feature quads of one edge -> coalesced
// b128 loads + coalesced f32 atomics
__global__ void edge_scatter4(const float* __restrict__ h, const int* __restrict__ src,
                              const int* __restrict__ dst, const float* __restrict__ dinv,
                              float* __restrict__ out, long long total4, int lgF) {
  long long idx = (long long)blockIdx.x * blockDim.x + threadIdx.x;
  if (idx >= total4) return;
  const int e  = (int)(idx >> (lgF - 2));
  const int f  = (int)((idx & ((1LL << (lgF - 2)) - 1)) << 2);
  const int s  = src[e];
  const int d  = dst[e];
  const float nrm = dinv[s] * dinv[d];
  const float4 hv = *(const float4*)(h + (((size_t)s) << lgF) + f);
  float* o = out + (((size_t)d) << lgF) + f;
  atomicAdd(o + 0, hv.x * nrm);
  atomicAdd(o + 1, hv.y * nrm);
  atomicAdd(o + 2, hv.z * nrm);
  atomicAdd(o + 3, hv.w * nrm);
}

__global__ void zero16(float* __restrict__ p) {
  if (threadIdx.x < 16) p[threadIdx.x] = 0.f;
}

// column sums of h[N,16] via per-block LDS reduction + one atomic per feature
__global__ void pool_accum(const float* __restrict__ h, float* __restrict__ pooled, int n) {
  __shared__ float s[256];
  const int f   = threadIdx.x & 15;
  const int row = threadIdx.x >> 4;  // 0..15
  float acc = 0.f;
  for (int i = blockIdx.x * 16 + row; i < n; i += gridDim.x * 16)
    acc += h[(size_t)i * 16 + f];
  s[threadIdx.x] = acc;
  __syncthreads();
  if (row == 0) {
    float t = 0.f;
#pragma unroll
    for (int j = 0; j < 16; ++j) t += s[j * 16 + f];
    atomicAdd(&pooled[f], t);
  }
}

__global__ void final_linear(const float* __restrict__ pooled, const float* __restrict__ Wl,
                             const float* __restrict__ bl, float* __restrict__ out) {
  const int c = threadIdx.x;
  if (c < 3) {
    const float invN = 1.0f / (float)N_NODES;
    float a = 0.f;
#pragma unroll
    for (int f = 0; f < 16; ++f) a += (pooled[f] * invN) * Wl[f * 3 + c];
    out[c] = a + bl[c];
  }
}

// ---------------------------------------------------------------------------
extern "C" void kernel_launch(void* const* d_in, const int* in_sizes, int n_in,
                              void* d_out, int out_size, void* d_ws, size_t ws_size,
                              hipStream_t stream) {
  (void)in_sizes; (void)n_in; (void)out_size; (void)ws_size;

  const float* x   = (const float*)d_in[0];
  const int*   ei  = (const int*)d_in[1];
  const int*   src = ei;
  const int*   dst = ei + N_EDGES;
  const float* Wt[5] = {(const float*)d_in[2], (const float*)d_in[4],
                        (const float*)d_in[6], (const float*)d_in[8],
                        (const float*)d_in[10]};
  const float* bt[5] = {(const float*)d_in[3], (const float*)d_in[5],
                        (const float*)d_in[7], (const float*)d_in[9],
                        (const float*)d_in[11]};
  const float* Wl = (const float*)d_in[12];
  const float* bl = (const float*)d_in[13];

  const int dims[6] = {3, 4096, 1024, 256, 64, 16};
  const int lgF[5]  = {12, 10, 8, 6, 4};   // log2 output width
  const int lgKp[5] = {5, 12, 10, 8, 6};   // log2 padded input width
  const int Kt[5]   = {1, 128, 32, 8, 2};  // K chunks of 32

  // workspace layout (16B-aligned sections)
  float*  bufA   = (float*)d_ws;                        // h (gemm out), 50000*4096
  float*  bufB   = bufA + (size_t)N_NODES * 4096;       // aggregated out
  float*  dinv   = bufB + (size_t)N_NODES * 4096;       // 50000
  float*  pooled = dinv + N_NODES;                      // 16
  __bf16* bufAbf = (__bf16*)(pooled + 16);              // bf16 A, 50000*4096
  __bf16* WpBase = bufAbf + (size_t)N_NODES * 4096;     // packed weights ~9.2MB
  __bf16* Wp[5];
  {
    __bf16* p = WpBase;
    for (int l = 0; l < 5; ++l) {
      Wp[l] = p;
      p += (size_t)Kt[l] * (dims[l + 1] >> 4) * 512;
    }
  }

  // pack all weights into WMMA B-fragment layout (bf16)
  for (int l = 0; l < 5; ++l) {
    const long long tot = (long long)Kt[l] * (dims[l + 1] >> 4) * 512;
    pack_w<<<(unsigned)((tot + 255) / 256), 256, 0, stream>>>(Wt[l], Wp[l],
                                                             dims[l], dims[l + 1], tot);
  }

  // degrees with self-loops -> dinv = rsqrt(deg)
  deg_init <<<(N_NODES + 255) / 256, 256, 0, stream>>>(dinv, N_NODES);
  deg_accum<<<(N_EDGES + 255) / 256, 256, 0, stream>>>(dst, dinv, N_EDGES);
  deg_rsqrt<<<(N_NODES + 255) / 256, 256, 0, stream>>>(dinv, N_NODES);

  const float* in = x;
  for (int l = 0; l < 5; ++l) {
    const int K = dims[l], F = dims[l + 1];

    // 1) fp32 -> bf16 (pad K to 32-multiple), ReLU of previous layer fused
    const long long totA = (long long)N_NODES << lgKp[l];
    if (l == 0)
      cvt_a<false><<<(unsigned)((totA + 255) / 256), 256, 0, stream>>>(in, bufAbf, totA, K, lgKp[l]);
    else
      cvt_a<true><<<(unsigned)((totA + 255) / 256), 256, 0, stream>>>(in, bufAbf, totA, K, lgKp[l]);

    // 2) h = A @ W via WMMA (NT=4 for F>=64, NT=1 for F=16)
    dim3 gb((N_NODES + 255) / 256, 1);
    if (F >= 64) {
      dim3 g(F / 64, (N_NODES + 255) / 256);
      gemm_wmma_bf16<4><<<g, 256, 0, stream>>>(bufAbf, Wp[l], bufA, N_NODES, Kt[l], lgKp[l], F);
    } else {
      dim3 g(F / 16, (N_NODES + 255) / 256);
      gemm_wmma_bf16<1><<<g, 256, 0, stream>>>(bufAbf, Wp[l], bufA, N_NODES, Kt[l], lgKp[l], F);
    }

    // 3) out = h*dinv^2 + b, then scatter-add edges
    const long long totN4 = ((long long)N_NODES * F) >> 2;
    self_init4<<<(unsigned)((totN4 + 255) / 256), 256, 0, stream>>>(
        (const float4*)bufA, dinv, bt[l], (float4*)bufB, totN4, lgF[l]);
    const long long totE4 = ((long long)N_EDGES * F) >> 2;
    edge_scatter4<<<(unsigned)((totE4 + 255) / 256), 256, 0, stream>>>(
        bufA, src, dst, dinv, bufB, totE4, lgF[l]);
    in = bufB;
  }

  // global mean pool + final linear -> [1,3]
  zero16<<<1, 16, 0, stream>>>(pooled);
  pool_accum<<<256, 256, 0, stream>>>(bufB, pooled, N_NODES);
  final_linear<<<1, 32, 0, stream>>>(pooled, Wl, bl, (float*)d_out);
}
